// Attention_49177375539704
// MI455X (gfx1250) — matrix operand order repaired and checked by reading, not compiled, not run on hardware
//
#include <hip/hip_runtime.h>

// ---------------------------------------------------------------------------
// Causal multi-head attention, bf16 WMMA pipeline for gfx1250 (MI455X).
//   x:[2,2048,1024] f32 -> out:[2,2048,1024] f32, HEADS=16, d=64
// Compute-bound (~52 GFLOP vs ~100 MB, all L2-resident) -> bf16 WMMA
// 16x16x32 everywhere, f32 accumulation, f32 softmax stats.
// Workspace layout (ushort/bf16 elements), total 48 MB:
//   xb 4M | WqT 1M | WkvT 2M | WoutT 1M | Qm 4M | Km 4M | Vt 4M | Om 4M
// ---------------------------------------------------------------------------

typedef __attribute__((ext_vector_type(16))) __bf16 v16bf;
typedef __attribute__((ext_vector_type(8)))  __bf16 v8bf;
typedef __attribute__((ext_vector_type(8)))  float  v8f;
typedef __attribute__((ext_vector_type(8)))  unsigned short us8;

__device__ __forceinline__ unsigned short f32_to_bf16_rne(float f) {
  unsigned u = __builtin_bit_cast(unsigned, f);
  u += 0x7fffu + ((u >> 16) & 1u);
  return (unsigned short)(u >> 16);
}

__device__ __forceinline__ v16bf cat16(v8bf lo, v8bf hi) {
  v16bf r;
#pragma unroll
  for (int i = 0; i < 8; ++i) { r[i] = lo[i]; r[i + 8] = hi[i]; }
  return r;
}

// ----------------------------- prep kernels --------------------------------

__global__ void f32_to_bf16_kernel(const float* __restrict__ in,
                                   unsigned short* __restrict__ out, int n) {
  for (int i = blockIdx.x * blockDim.x + threadIdx.x; i < n;
       i += gridDim.x * blockDim.x)
    out[i] = f32_to_bf16_rne(in[i]);
}

// in: f32 [K][N] row-major  ->  out: bf16 [N][K] row-major
__global__ void transpose_to_bf16_kernel(const float* __restrict__ in,
                                         unsigned short* __restrict__ out,
                                         int K, int N) {
  int total = K * N;
  for (int i = blockIdx.x * blockDim.x + threadIdx.x; i < total;
       i += gridDim.x * blockDim.x) {
    int k = i / N, n = i - k * N;
    out[(size_t)n * K + k] = f32_to_bf16_rne(in[i]);
  }
}

// ------------------------------- GEMM --------------------------------------
// C[M,N] = (A[M,K] @ B + bias) * alpha,  B given transposed as BT[N][K] bf16.
// One wave -> 64x64 C tile: 4x4 WMMA 16x16x32 bf16 accum tiles
// (16 WMMAs per 16 b128 loads per 32-K step).
// MODE 0: bf16 store [M][ldc]      (Q, K projections)
// MODE 1: bf16 store transposed per head: Vt[(b*H+h)*64+dd][nseq] (V proj)
// MODE 2: f32  store [M][ldc]      (output projection)
template <int MODE>
__global__ void __launch_bounds__(32)
gemm_wmma_kernel(const unsigned short* __restrict__ A, int lda,
                 const unsigned short* __restrict__ BT, int ldb,
                 const float* __restrict__ bias, float alpha,
                 unsigned short* __restrict__ Cb, float* __restrict__ Cf,
                 int ldc, int Kdim, int nseq, int H) {
  const int lane = threadIdx.x & 31;
  const int hl = lane >> 4;    // half-wave select
  const int ln = lane & 15;
  const int M0 = blockIdx.x * 64;
  const int N0 = blockIdx.y * 64;

  v8f acc[4][4] = {};
  for (int k0 = 0; k0 < Kdim; k0 += 32) {
    v16bf af[4], bfr[4];
#pragma unroll
    for (int mi = 0; mi < 4; ++mi) {
      // A fragment: lane holds row M0+mi*16+ln; K chunks {8*hl..} u {16+8*hl..}
      const unsigned short* p =
          A + (size_t)(M0 + mi * 16 + ln) * lda + k0 + hl * 8;
      af[mi] = cat16(*(const v8bf*)p, *(const v8bf*)(p + 16));
      __builtin_prefetch(p + 32, 0, 3);  // global_prefetch_b8 for next K step
    }
#pragma unroll
    for (int nj = 0; nj < 4; ++nj) {
      // B fragment: lane holds column N0+nj*16+ln; K chunks {8*hl..} u {16+8*hl..}
      const unsigned short* p =
          BT + (size_t)(N0 + nj * 16 + ln) * ldb + k0 + hl * 8;
      bfr[nj] = cat16(*(const v8bf*)p, *(const v8bf*)(p + 16));
    }
#pragma unroll
    for (int mi = 0; mi < 4; ++mi)
#pragma unroll
      for (int nj = 0; nj < 4; ++nj)
        acc[mi][nj] = __builtin_amdgcn_wmma_f32_16x16x32_bf16(
            false, af[mi], false, bfr[nj], (short)0, acc[mi][nj], false, false);
  }

#pragma unroll
  for (int nj = 0; nj < 4; ++nj) {
    const int n = N0 + nj * 16 + ln;
    const float bv = bias[n];
#pragma unroll
    for (int mi = 0; mi < 4; ++mi) {
      const int mbase = M0 + mi * 16 + hl * 8;  // C layout: VGPR r -> M=mbase+r
      if (MODE == 0) {
#pragma unroll
        for (int r = 0; r < 8; ++r)
          Cb[(size_t)(mbase + r) * ldc + n] =
              f32_to_bf16_rne((acc[mi][nj][r] + bv) * alpha);
      } else if (MODE == 2) {
#pragma unroll
        for (int r = 0; r < 8; ++r)
          Cf[(size_t)(mbase + r) * ldc + n] = (acc[mi][nj][r] + bv) * alpha;
      } else {  // MODE 1: per-head transposed V store, packed b128
        const int hh = n >> 6, dd = n & 63;
        const int bb = mbase / nseq, tb = mbase % nseq;
        us8 pk;
#pragma unroll
        for (int r = 0; r < 8; ++r)
          pk[r] = f32_to_bf16_rne((acc[mi][nj][r] + bv) * alpha);
        *(us8*)(Cb + ((size_t)(bb * H + hh) * 64 + dd) * (size_t)nseq + tb) = pk;
      }
    }
  }
}

// --------------------------- flash attention -------------------------------
// One wave handles 16 query rows of one (b,h). Transposed-score scheme:
//   St = K @ Q^T   (keys on M axis -> softmax reduction mostly in-lane)
//   Ot = V^T @ P^T (P^T B-fragment built from St regs with cross-half shfl)
// The causal mask can only trigger in the final 32-key tile (tiles are
// aligned), so the steady-state loop runs MASKED=false with zero mask VALU.

template <bool MASKED>
__device__ __forceinline__ void attn_key_tile(
    const unsigned short* __restrict__ kbase,
    const unsigned short* __restrict__ vbase, const v16bf* qf, int kb, int row,
    int hl, int ln, int nseq, int dim, float& mrun, float& lrun, v8f* ot) {
  v8f st[2] = {};
#pragma unroll
  for (int mt = 0; mt < 2; ++mt) {
    const unsigned short* krow = kbase + (size_t)(kb + mt * 16 + ln) * dim;
#pragma unroll
    for (int ks = 0; ks < 2; ++ks) {
      const unsigned short* p = krow + ks * 32 + hl * 8;
      v16bf kf = cat16(*(const v8bf*)p, *(const v8bf*)(p + 16));
      st[mt] = __builtin_amdgcn_wmma_f32_16x16x32_bf16(
          false, kf, false, qf[ks], (short)0, st[mt], false, false);
    }
  }

  // (optional) causal mask + running max over keys (M axis of St)
  float s[2][8];
  float mx = -1e30f;
#pragma unroll
  for (int mt = 0; mt < 2; ++mt)
#pragma unroll
    for (int r = 0; r < 8; ++r) {
      float v = st[mt][r];
      if (MASKED) {
        const int key = kb + mt * 16 + hl * 8 + r;
        if (key > row) v = -1e30f;
      }
      s[mt][r] = v;
      mx = fmaxf(mx, v);
    }
  mx = fmaxf(mx, __shfl_xor(mx, 16, 32));
  const float mnew = fmaxf(mrun, mx);
  const float corr = __expf(mrun - mnew);
  float psum = 0.0f;
#pragma unroll
  for (int mt = 0; mt < 2; ++mt)
#pragma unroll
    for (int r = 0; r < 8; ++r) {
      const float e = __expf(s[mt][r] - mnew);
      s[mt][r] = e;
      psum += e;
    }
  psum += __shfl_xor(psum, 16, 32);
  lrun = lrun * corr + psum;
  mrun = mnew;
#pragma unroll
  for (int dt = 0; dt < 4; ++dt)
#pragma unroll
    for (int r = 0; r < 8; ++r) ot[dt][r] *= corr;

  // P^T B-fragment: lane ln = query col; needs keys 16*hl + 0..15.
  // Own regs cover keys {8*hl+r, 16+8*hl+r}; fetch the other 8 via xor-16.
  union { v16bf v; unsigned short u[16]; } pf;
#pragma unroll
  for (int j = 0; j < 8; ++j) {
    const float o0 = __shfl_xor(s[0][j], 16, 32);
    const float o1 = __shfl_xor(s[1][j], 16, 32);
    const float elo = hl ? o1 : s[0][j];  // key 16*hl + j
    const float ehi = hl ? s[1][j] : o0;  // key 16*hl + 8 + j
    pf.u[j] = f32_to_bf16_rne(elo);
    pf.u[j + 8] = f32_to_bf16_rne(ehi);
  }

#pragma unroll
  for (int dt = 0; dt < 4; ++dt) {
    // V^T A-fragment: lane holds d-row dt*16+ln, key chunks from Vt (contig)
    const unsigned short* p =
        vbase + (size_t)(dt * 16 + ln) * nseq + kb + hl * 16;
    v16bf vf = cat16(*(const v8bf*)p, *(const v8bf*)(p + 8));
    ot[dt] = __builtin_amdgcn_wmma_f32_16x16x32_bf16(
        false, vf, false, pf.v, (short)0, ot[dt], false, false);
  }
}

__global__ void __launch_bounds__(32)
flash_attn_kernel(const unsigned short* __restrict__ Qm,
                  const unsigned short* __restrict__ Km,
                  const unsigned short* __restrict__ Vt,
                  unsigned short* __restrict__ Om, int nseq, int H) {
  const int lane = threadIdx.x & 31;
  const int hl = lane >> 4;
  const int ln = lane & 15;
  const int rb = blockIdx.x * 16;  // query row tile base
  const int h = blockIdx.y;
  const int b = blockIdx.z;
  const int dim = H * 64;

  // Q fragments (B operand of St): lane ln holds query row rb+ln
  const unsigned short* qrow = Qm + (size_t)(b * nseq + rb + ln) * dim + h * 64;
  v16bf qf[2];
#pragma unroll
  for (int ks = 0; ks < 2; ++ks) {
    const unsigned short* p = qrow + ks * 32 + hl * 8;
    qf[ks] = cat16(*(const v8bf*)p, *(const v8bf*)(p + 16));
  }

  const unsigned short* kbase = Km + (size_t)(b * nseq) * dim + h * 64;
  const unsigned short* vbase = Vt + (size_t)(b * H + h) * 64 * (size_t)nseq;

  float mrun = -1e30f, lrun = 0.0f;
  v8f ot[4] = {};  // O^T accum: 4 d-tiles x (16 d-rows x 16 query rows)
  const int row = rb + ln;

  // Last 32-aligned key tile containing key rb+15 is the only one that can
  // cross the diagonal; everything before it is fully unmasked.
  const int kb_last = ((rb + 15) >> 5) << 5;
  for (int kb = 0; kb < kb_last; kb += 32)
    attn_key_tile<false>(kbase, vbase, qf, kb, row, hl, ln, nseq, dim, mrun,
                         lrun, ot);
  attn_key_tile<true>(kbase, vbase, qf, kb_last, row, hl, ln, nseq, dim, mrun,
                      lrun, ot);

  // Ot lane holds O[query row ln][d = dt*16 + 8*hl + r] -> packed b128 store
  const float inv = 1.0f / lrun;
  unsigned short* orow = Om + (size_t)(b * nseq + rb + ln) * dim + h * 64;
#pragma unroll
  for (int dt = 0; dt < 4; ++dt) {
    us8 pk;
#pragma unroll
    for (int r = 0; r < 8; ++r) pk[r] = f32_to_bf16_rne(ot[dt][r] * inv);
    *(us8*)(orow + dt * 16 + hl * 8) = pk;
  }
}

// ------------------------------ launcher -----------------------------------

extern "C" void kernel_launch(void* const* d_in, const int* in_sizes, int n_in,
                              void* d_out, int out_size, void* d_ws,
                              size_t ws_size, hipStream_t stream) {
  const float* x    = (const float*)d_in[0];
  const float* Wq   = (const float*)d_in[1];
  const float* bq   = (const float*)d_in[2];
  const float* Wkv  = (const float*)d_in[3];
  const float* bkv  = (const float*)d_in[4];
  const float* Wout = (const float*)d_in[5];
  const float* bout = (const float*)d_in[6];
  float* out = (float*)d_out;

  const int B = 2, NSEQ = 2048, DIM = 1024, H = 16;
  const int M = B * NSEQ;      // 4096 tokens
  const float scale = 0.125f;  // d^-0.5, folded into Q projection

  unsigned short* ws = (unsigned short*)d_ws;
  unsigned short* xb    = ws;                                  // [M][DIM]
  unsigned short* WqT   = xb + (size_t)M * DIM;                // [DIM][DIM]
  unsigned short* WkvT  = WqT + (size_t)DIM * DIM;             // [2*DIM][DIM]
  unsigned short* WoutT = WkvT + (size_t)DIM * 2 * DIM;        // [DIM][DIM]
  unsigned short* Qm    = WoutT + (size_t)DIM * DIM;           // [M][DIM]
  unsigned short* Km    = Qm + (size_t)M * DIM;                // [M][DIM]
  unsigned short* Vt    = Km + (size_t)M * DIM;                // [B*H*64][NSEQ]
  unsigned short* Om    = Vt + (size_t)M * DIM;                // [M][DIM]

  f32_to_bf16_kernel<<<2048, 256, 0, stream>>>(x, xb, M * DIM);
  transpose_to_bf16_kernel<<<1024, 256, 0, stream>>>(Wq, WqT, DIM, DIM);
  transpose_to_bf16_kernel<<<2048, 256, 0, stream>>>(Wkv, WkvT, DIM, 2 * DIM);
  transpose_to_bf16_kernel<<<1024, 256, 0, stream>>>(Wout, WoutT, DIM, DIM);

  dim3 g(M / 64, DIM / 64);  // 64 x 16 wave-tiles
  // Q = (x @ Wq + bq) * scale
  gemm_wmma_kernel<0><<<g, 32, 0, stream>>>(xb, DIM, WqT, DIM, bq, scale,
                                            Qm, nullptr, DIM, DIM, NSEQ, H);
  // K = x @ Wkv[:, :DIM] + bkv[:DIM]
  gemm_wmma_kernel<0><<<g, 32, 0, stream>>>(xb, DIM, WkvT, DIM, bkv, 1.0f,
                                            Km, nullptr, DIM, DIM, NSEQ, H);
  // V (stored transposed per head): x @ Wkv[:, DIM:] + bkv[DIM:]
  gemm_wmma_kernel<1><<<g, 32, 0, stream>>>(xb, DIM, WkvT + (size_t)DIM * DIM,
                                            DIM, bkv + DIM, 1.0f, Vt, nullptr,
                                            0, DIM, NSEQ, H);

  flash_attn_kernel<<<dim3(NSEQ / 16, H, B), 32, 0, stream>>>(Qm, Km, Vt, Om,
                                                              NSEQ, H);

  // out = Om @ Wout + bout  (f32 output)
  gemm_wmma_kernel<2><<<g, 32, 0, stream>>>(Om, DIM, WoutT, DIM, bout, 1.0f,
                                            nullptr, out, DIM, DIM, NSEQ, H);
}
